// VariationalSDE_85985245266166
// MI455X (gfx1250) — compile-verified, matrix operand order checked by reading, not run on hardware
//
#include <hip/hip_runtime.h>
#include <hip/hip_bf16.h>
#include <math.h>

typedef __attribute__((ext_vector_type(16))) _Float16 v16h;
typedef __attribute__((ext_vector_type(8)))  float    v8f;

#if defined(__has_builtin)
#if __has_builtin(__builtin_amdgcn_tanhf)
#define FAST_TANH(x) __builtin_amdgcn_tanhf(x)
#endif
#endif
#ifndef FAST_TANH
#define FAST_TANH(x) tanhf(x)
#endif

__global__ __launch_bounds__(256)
void vsde_wmma_kernel(const float* __restrict__ z0_raw,
                      const float* __restrict__ dW_raw,
                      const float* __restrict__ W1,   // (2,64) row-major
                      const float* __restrict__ b1,   // (64)
                      const float* __restrict__ W2,   // (64,1)
                      const float* __restrict__ b2p,  // (1)
                      float* __restrict__ out,        // [50*N z_stack | N kldiv]
                      int N)
{
    const int tid  = blockIdx.x * blockDim.x + threadIdx.x;
    const int lane = threadIdx.x & 31;
    const int half = (lane >> 4) & 1;        // which half of the wave
    const bool lo  = (lane < 16);
    const int idx  = (tid < N) ? tid : (N - 1);   // clamp loads; EXEC stays all-1s for WMMA

    // ---- static WMMA operands held in registers ----
    // A[tile]: 16x32 f16, row h = hidden unit tile*16+h, K0=W1[0,h], K1=W1[1,h]
    // Cb[tile]: f32 accumulator pre-load = b1 in D layout (exact bias path)
    // w2v[tile][r]: W2 in D layout for the layer-2 dot product
    v16h A[4]; v8f Cb[4]; float w2v[4][8];
#pragma unroll
    for (int t4 = 0; t4 < 4; ++t4) {
#pragma unroll
        for (int i = 0; i < 16; ++i) A[t4][i] = (_Float16)0.f;
        const int j = t4 * 16 + (lane & 15);
        const _Float16 a0 = (_Float16)W1[j];        // x-weight
        const _Float16 a1 = (_Float16)W1[64 + j];   // t-weight
        A[t4][0] = lo ? a0 : (_Float16)0.f;
        A[t4][1] = lo ? a1 : (_Float16)0.f;
#pragma unroll
        for (int r = 0; r < 8; ++r) {
            const int h = t4 * 16 + r + 8 * half;
            Cb[t4][r]  = b1[h];
            w2v[t4][r] = W2[h];
        }
    }
    const float b2 = b2p[0];

    // f(t, x) for this lane's sample; whole wave cooperates (32 samples / wave)
    auto eval_f = [&](float x, float t0) -> float {
        const _Float16 th = (_Float16)t0;
        const float xsw = __shfl_xor(x, 16, 32);    // other half's samples for group 1
        float p0 = 0.f, p1 = 0.f;
#pragma unroll
        for (int g = 0; g < 2; ++g) {
            const float xg = (g == 0) ? x : xsw;
            v16h B;
#pragma unroll
            for (int i = 0; i < 16; ++i) B[i] = (_Float16)0.f;
            B[0] = lo ? (_Float16)xg : (_Float16)0.f;  // K=0 row: x_n
            B[1] = lo ? th           : (_Float16)0.f;  // K=1 row: t
            float acc = 0.f;
#pragma unroll
            for (int t4 = 0; t4 < 4; ++t4) {
                // pre[h, s] = a_h * x_s + d_h * t + b1_h   (bias exact via f32 C)
                v8f D = __builtin_amdgcn_wmma_f32_16x16x32_f16(
                    false, A[t4], false, B, (short)0, Cb[t4], false, false);
#pragma unroll
                for (int r = 0; r < 8; ++r)
                    acc = fmaf(FAST_TANH(D[r]), w2v[t4][r], acc);
            }
            if (g == 0) p0 = acc; else p1 = acc;
        }
        // sum the two hidden-halves (lanes l and l^16 hold complementary h's)
        const float f0 = p0 + __shfl_xor(p0, 16, 32);  // sample (lane&15)
        const float f1 = p1 + __shfl_xor(p1, 16, 32);  // sample (lane&15)+16
        return (lo ? f0 : f1) + b2;
    };

    float z = z0_raw[idx] * 3.0f;

    // ---- ODE phase: 49 Euler steps on t = linspace(0, 0.3, 50) ----
    const float dt_o = 0.3f / 49.0f;
#pragma unroll 1
    for (int i = 0; i < 49; ++i) {
        const float t0 = (float)i * dt_o;
        z += eval_f(z, t0) * dt_o;
    }
    if (tid < N) out[tid] = z;                       // z_stack[0] = z_cut

    // ---- SDE phase: 49 EM steps on t = linspace(0.3, 3.0, 50) + KL ----
    const float dt_s  = 2.7f / 49.0f;
    const float sq_dt = sqrtf(dt_s);
    const float g_c   = 1.4142135623730951f;         // sqrt(2)
    const float inv_g = 0.7071067811865476f;
    float kl = 0.f;
#pragma unroll 1
    for (int i = 0; i < 49; ++i) {
        const float t0 = 0.3f + (float)i * dt_s;
        if (i + 1 < 49)                               // gfx1250 global_prefetch path
            __builtin_prefetch(dW_raw + (size_t)(i + 1) * N + idx, 0, 1);
        const float dW = dW_raw[(size_t)i * N + idx] * sq_dt;
        const float f  = eval_f(z, t0);
        z = z + f * dt_s + g_c * dW;
        // f_prior = -0.1*z_new ; u = (f - f_prior)/g  (g >= 0.001 always)
        const float u = (f + 0.1f * z) * inv_g;
        kl += dt_s * 0.5f * u * u;
        if (tid < N) out[(size_t)(i + 1) * N + tid] = z;
    }
    if (tid < N) out[(size_t)50 * N + tid] = kl;     // kldiv
}

extern "C" void kernel_launch(void* const* d_in, const int* in_sizes, int n_in,
                              void* d_out, int out_size, void* d_ws, size_t ws_size,
                              hipStream_t stream) {
    // inputs: [0]=nsamples(int,1) [1]=z0_raw(N) [2]=dW_raw(49*N)
    //         [3]=W1(128) [4]=b1(64) [5]=W2(64) [6]=b2(1)
    const float* z0 = (const float*)d_in[1];
    const float* dW = (const float*)d_in[2];
    const float* W1 = (const float*)d_in[3];
    const float* b1 = (const float*)d_in[4];
    const float* W2 = (const float*)d_in[5];
    const float* b2 = (const float*)d_in[6];
    float* out = (float*)d_out;
    const int N = in_sizes[1];
    const int threads = 256;
    const int blocks = (N + threads - 1) / threads;
    hipLaunchKernelGGL(vsde_wmma_kernel, dim3(blocks), dim3(threads), 0, stream,
                       z0, dW, W1, b1, W2, b2, out, N);
}
